// vrnn_58445914964278
// MI455X (gfx1250) — matrix operand order repaired and checked by reading, not compiled, not run on hardware
//
#include <hip/hip_runtime.h>
#include <math.h>
#include <stdint.h>

// ---------------------------------------------------------------------------
// VRNN on MI455X (gfx1250): bf16 WMMA GEMMs with double-buffered
// GLOBAL_LOAD_ASYNC_TO_LDS pipelines (ASYNCcnt via inline asm), h-independent
// work hoisted out of the scan and batched over all S*B tokens; recurrent
// phase runs 128 dependent steps whose weights (~14MB bf16) stay in L2.
// ---------------------------------------------------------------------------

typedef __attribute__((ext_vector_type(16))) __bf16 v16bf;
typedef __attribute__((ext_vector_type(8)))  __bf16 v8bf;
typedef __attribute__((ext_vector_type(8)))  float  v8f;

static constexpr int S = 128, B = 256, E = 1024, H = 1024, Z = 1024;
static constexpr int T = S * B;      // 32768 tokens
static constexpr int FH = 4 * H;     // 4096

// ---- CDNA5 async global->LDS path (device pass only; host parses fallback)
#if defined(__gfx1250__)
#  define USE_ASYNC_LDS 1
#else
#  define USE_ASYNC_LDS 0
#endif

#if USE_ASYNC_LDS
// s_wait_asynccnt <= n  (ASYNCcnt tracks async-to-LDS completions, in order)
#  define ASYNC_WAIT(n) asm volatile("s_wait_asynccnt %0" :: "i"(n) : "memory")
// VDST = VGPR with LDS byte address (low 32 bits of generic LDS pointer),
// VADDR = 64-bit global address, "off" = no SADDR.
__device__ __forceinline__ void async_load_b128(const void* gaddr, void* lds) {
    asm volatile("global_load_async_to_lds_b128 %0, %1, off"
                 :: "v"((unsigned)(uintptr_t)lds),
                    "v"((unsigned long long)(uintptr_t)gaddr)
                 : "memory");
}
#endif

union FragBf { v16bf v; v8bf h[2]; };

__device__ __forceinline__ __bf16 f2bf(float x) {
    union { float f; unsigned u; } a; a.f = x;
    unsigned r = a.u + 0x7FFFu + ((a.u >> 16) & 1u);   // round-to-nearest-even
    union { unsigned short s; __bf16 b; } o; o.s = (unsigned short)(r >> 16);
    return o.b;
}

__device__ __forceinline__ float sigmoidf_(float x) { return 1.0f / (1.0f + __expf(-x)); }

// ------------------------- elementwise helpers -----------------------------

__global__ void f32_to_bf16_kernel(const float* __restrict__ in, __bf16* __restrict__ out, size_t n) {
    size_t stride = (size_t)gridDim.x * blockDim.x;
    for (size_t i = blockIdx.x * (size_t)blockDim.x + threadIdx.x; i < n; i += stride)
        out[i] = f2bf(in[i]);
}

__global__ void zero_u32_kernel(unsigned* __restrict__ p, size_t n) {
    size_t stride = (size_t)gridDim.x * blockDim.x;
    for (size_t i = blockIdx.x * (size_t)blockDim.x + threadIdx.x; i < n; i += stride)
        p[i] = 0u;
}

__global__ void bias_sum_kernel(const float* __restrict__ a, const float* __restrict__ b,
                                float* __restrict__ out, int n) {
    int i = blockIdx.x * blockDim.x + threadIdx.x;
    if (i < n) out[i] = a[i] + b[i];
}

// ------------------------------ bf16 GEMM ----------------------------------
// C[M,N] = A[M,K] * W[N,K]^T (+ bias[N]) (+ Cadd[M,N] for MODE 2)
// MODE 0: f32 out   MODE 1: bf16 out   MODE 2: f32 out with Cadd accumulate
// Requires M,N multiples of 128 and K multiple of 32 (true for all calls).
template <int MODE>
__global__ __launch_bounds__(128) void gemm_bf16_kernel(
    const __bf16* __restrict__ A, const __bf16* __restrict__ W,
    void* __restrict__ Cout, const float* __restrict__ bias,
    const float* __restrict__ Cadd, int M, int N, int K)
{
    constexpr int TM = 128, TN = 128, TK = 32;
    constexpr int LDA = TK + 8;                       // padded LDS stride (bf16 elems)
    __shared__ __align__(16) __bf16 Als[2][TM * LDA];
    __shared__ __align__(16) __bf16 Bls[2][TN * LDA];

    const int nTilesN = N / TN;
    const int bm = blockIdx.x / nTilesN;
    const int bn = blockIdx.x % nTilesN;
    const int tid  = threadIdx.x;
    const int lane = tid & 31;
    const int wave = tid >> 5;                        // 4 waves per WG
    const int wm = (wave >> 1) * 64;                  // 2x2 wave grid, 64x64 each
    const int wn = (wave & 1) * 64;
    const int lhalf = lane >> 4;                      // 0/1
    const int l15   = lane & 15;

    v8f acc[4][4] = {};

    const __bf16* Arow = A + (size_t)(bm * TM + tid) * K;  // thread t owns row t
    const __bf16* Wrow = W + (size_t)(bn * TN + tid) * K;

    auto compute_tile = [&](int buf) {
        v16bf afrag[4], bfrag[4];
        // A 16x32 fragment: lane-half h holds K = {8h..8h+7} U {16+8h..16+8h+7}
        for (int i = 0; i < 4; i++) {
            const __bf16* base = &Als[buf][(wm + i * 16 + l15) * LDA];
            FragBf f;
            f.h[0] = *(const v8bf*)(base + lhalf * 8);
            f.h[1] = *(const v8bf*)(base + 16 + lhalf * 8);
            afrag[i] = f.v;
        }
        // B 32x16 fragment: lane-half h holds K = {8h..8h+7} U {16+8h..16+8h+7}
        for (int j = 0; j < 4; j++) {
            const __bf16* base = &Bls[buf][(wn + j * 16 + l15) * LDA];
            FragBf f;
            f.h[0] = *(const v8bf*)(base + lhalf * 8);
            f.h[1] = *(const v8bf*)(base + 16 + lhalf * 8);
            bfrag[j] = f.v;
        }
        for (int i = 0; i < 4; i++)
            for (int j = 0; j < 4; j++)
                acc[i][j] = __builtin_amdgcn_wmma_f32_16x16x32_bf16(
                    false, afrag[i], false, bfrag[j], (short)0, acc[i][j], false, false);
    };

#if USE_ASYNC_LDS
    // 2-deep pipeline on the CDNA5 async-to-LDS engine: tile fill overlaps
    // the 16 WMMAs of the previous tile; ASYNCcnt tracks completion in order.
    auto issue_tile = [&](int buf, int k0) {
        const __bf16* ga = Arow + k0;
        const __bf16* gw = Wrow + k0;
        __bf16* la = &Als[buf][tid * LDA];
        __bf16* lw = &Bls[buf][tid * LDA];
        #pragma unroll
        for (int ch = 0; ch < 4; ++ch) {              // 4 x 16B per row each
            async_load_b128(ga + ch * 8, la + ch * 8);
            async_load_b128(gw + ch * 8, lw + ch * 8);
        }
    };
    const int nk = K / TK;
    issue_tile(0, 0);                                  // 8 async ops in flight
    for (int kt = 0; kt < nk; ++kt) {
        const int cur = kt & 1;
        if (kt + 1 < nk) {
            issue_tile(cur ^ 1, (kt + 1) * TK);        // 16 in flight
            ASYNC_WAIT(8);                             // tile kt done; kt+1 pending
        } else {
            ASYNC_WAIT(0);
        }
        __syncthreads();                               // LDS visible to all waves
        compute_tile(cur);
        __syncthreads();                               // all reads done before overwrite
    }
#else
    // fallback: synchronous VGPR staging (also what the host pass parses)
    for (int k0 = 0; k0 < K; k0 += TK) {
        const int4* sa = (const int4*)(Arow + k0);
        int4* da = (int4*)(&Als[0][tid * LDA]);
        da[0] = sa[0]; da[1] = sa[1]; da[2] = sa[2]; da[3] = sa[3];
        const int4* sw = (const int4*)(Wrow + k0);
        int4* dw = (int4*)(&Bls[0][tid * LDA]);
        dw[0] = sw[0]; dw[1] = sw[1]; dw[2] = sw[2]; dw[3] = sw[3];
        __syncthreads();
        compute_tile(0);
        __syncthreads();
    }
#endif

    // epilogue: C/D layout — VGPR r of lane L is (m = r + 8*(L>>4), n = L&15)
    const int row0 = bm * TM + wm + lhalf * 8;
    const int col0 = bn * TN + wn + l15;
    for (int i = 0; i < 4; i++)
        for (int j = 0; j < 4; j++) {
            const int col = col0 + j * 16;
            const float bv = bias ? bias[col] : 0.0f;
            for (int r = 0; r < 8; r++) {
                const size_t off = (size_t)(row0 + i * 16 + r) * N + col;
                float v = acc[i][j][r] + bv;
                if constexpr (MODE == 2) v += Cadd[off];
                if constexpr (MODE == 1) ((__bf16*)Cout)[off] = f2bf(v);
                else                     ((float*)Cout)[off]  = v;
            }
        }
}

// ------------------- z = mu + eps*sqrt(var), argmax over Z -----------------
// exp(0.5*log(var)) == sqrt(var); log applied separately for KL.
__global__ __launch_bounds__(256) void z_argmax_kernel(
    const float* __restrict__ qmu, const float* __restrict__ qvar,
    const float* __restrict__ eps, float* __restrict__ zout, int* __restrict__ idx)
{
    const int t = blockIdx.x;
    const size_t base = (size_t)t * Z;
    float best = -3.4e38f; int bi = 0;
    for (int zi = threadIdx.x; zi < Z; zi += 256) {
        float zv = qmu[base + zi] + eps[base + zi] * sqrtf(qvar[base + zi]);
        zout[base + zi] = zv;
        if (zv > best) { best = zv; bi = zi; }
    }
    __shared__ float sv[256];
    __shared__ int   si[256];
    sv[threadIdx.x] = best; si[threadIdx.x] = bi;
    __syncthreads();
    for (int s2 = 128; s2 > 0; s2 >>= 1) {
        if (threadIdx.x < (unsigned)s2) {
            float ov = sv[threadIdx.x + s2]; int oi = si[threadIdx.x + s2];
            if (ov > sv[threadIdx.x] || (ov == sv[threadIdx.x] && oi < si[threadIdx.x])) {
                sv[threadIdx.x] = ov; si[threadIdx.x] = oi;
            }
        }
        __syncthreads();
    }
    if (threadIdx.x == 0) idx[t] = si[0];
}

// ---------------------- embedding gather (bf16 rows) -----------------------
__global__ __launch_bounds__(128) void gather_kernel(
    const __bf16* __restrict__ temb, const int* __restrict__ idx, __bf16* __restrict__ X)
{
    const int t = blockIdx.x;
    const int4* src = (const int4*)(temb + (size_t)idx[t] * E);
    int4* dst = (int4*)(X + (size_t)t * E);
    dst[threadIdx.x] = src[threadIdx.x];               // 128 threads x 16B = E bf16
}

// ------------------------------- KL term -----------------------------------
__global__ __launch_bounds__(256) void kl_kernel(
    const float* __restrict__ qmu, const float* __restrict__ qvar,
    const float* __restrict__ pmu, const float* __restrict__ pvar,
    float* __restrict__ kl)
{
    const int b = blockIdx.x;
    const size_t base = (size_t)b * Z;
    float acc = 0.0f;
    for (int zi = threadIdx.x; zi < Z; zi += 256) {
        float qv = qvar[base + zi];                    // exp(q_logvar)
        float pv = pvar[base + zi];                    // exp(p_logvar)
        float dm = qmu[base + zi] - pmu[base + zi];
        acc += 0.5f * (logf(pv) - logf(qv) + (qv + dm * dm) / pv - 1.0f);
    }
    __shared__ float ss[256];
    ss[threadIdx.x] = acc;
    __syncthreads();
    for (int s2 = 128; s2 > 0; s2 >>= 1) {
        if (threadIdx.x < (unsigned)s2) ss[threadIdx.x] += ss[threadIdx.x + s2];
        __syncthreads();
    }
    if (threadIdx.x == 0) kl[b] = ss[0];
}

// ------------------------------ LSTM cell ----------------------------------
__global__ void lstm_kernel(
    const float* __restrict__ gates, float* __restrict__ c, __bf16* __restrict__ h_bf)
{
    const int i = blockIdx.x * blockDim.x + threadIdx.x;   // over B*H
    const int b = i >> 10, hh = i & (H - 1);
    const float* g = gates + (size_t)b * FH;
    const float ig = sigmoidf_(g[hh]);
    const float fg = sigmoidf_(g[H + hh]);
    const float gg = tanhf(g[2 * H + hh]);
    const float og = sigmoidf_(g[3 * H + hh]);
    const float cn = fg * c[i] + ig * gg;
    c[i] = cn;
    h_bf[i] = f2bf(og * tanhf(cn));
}

// ------------------------------- launcher ----------------------------------

extern "C" void kernel_launch(void* const* d_in, const int* in_sizes, int n_in,
                              void* d_out, int out_size, void* d_ws, size_t ws_size,
                              hipStream_t stream) {
    (void)in_sizes; (void)n_in; (void)out_size; (void)ws_size;

    const float* rnn_w  = (const float*)d_in[0];
    const float* eps    = (const float*)d_in[1];
    const float* W_q    = (const float*)d_in[2];
    const float* b_q    = (const float*)d_in[3];
    const float* W_qmu  = (const float*)d_in[4];
    const float* b_qmu  = (const float*)d_in[5];
    const float* W_qvar = (const float*)d_in[6];
    const float* b_qvar = (const float*)d_in[7];
    const float* W_p    = (const float*)d_in[8];
    const float* b_p    = (const float*)d_in[9];
    const float* W_pmu  = (const float*)d_in[10];
    const float* b_pmu  = (const float*)d_in[11];
    const float* W_pvar = (const float*)d_in[12];
    const float* b_pvar = (const float*)d_in[13];
    const float* temb   = (const float*)d_in[14];
    const float* W_ih   = (const float*)d_in[15];
    const float* W_hh   = (const float*)d_in[16];
    const float* b_ih   = (const float*)d_in[17];
    const float* b_hh   = (const float*)d_in[18];

    float* z_out  = (float*)d_out;                 // [S,B,Z]
    float* kl_out = (float*)d_out + (size_t)T * Z; // [S,B]

    // workspace carve-up (~1 GB total)
    char* ws = (char*)d_ws;
    size_t off = 0;
    auto alloc = [&](size_t bytes) -> void* {
        void* p = ws + off;
        off += (bytes + 255) & ~(size_t)255;
        return p;
    };
    __bf16* wq_bf    = (__bf16*)alloc((size_t)H * E * 2);
    __bf16* wqmu_bf  = (__bf16*)alloc((size_t)Z * H * 2);
    __bf16* wqvar_bf = (__bf16*)alloc((size_t)Z * H * 2);
    __bf16* wp_bf    = (__bf16*)alloc((size_t)H * H * 2);
    __bf16* wpmu_bf  = (__bf16*)alloc((size_t)Z * H * 2);
    __bf16* wpvar_bf = (__bf16*)alloc((size_t)Z * H * 2);
    __bf16* wih_bf   = (__bf16*)alloc((size_t)FH * E * 2);
    __bf16* whh_bf   = (__bf16*)alloc((size_t)FH * H * 2);
    __bf16* temb_bf  = (__bf16*)alloc((size_t)Z * E * 2);
    float*  bias_g   = (float*)alloc((size_t)FH * 4);
    __bf16* rnnw_bf  = (__bf16*)alloc((size_t)T * E * 2);
    __bf16* qi_bf    = (__bf16*)alloc((size_t)T * H * 2);
    float*  qmu      = (float*)alloc((size_t)T * Z * 4);
    float*  qvar     = (float*)alloc((size_t)T * Z * 4);
    int*    idx      = (int*)alloc((size_t)T * 4);
    __bf16* X_bf     = (__bf16*)alloc((size_t)T * E * 2);
    float*  gatesx   = (float*)alloc((size_t)T * FH * 4);
    __bf16* h_bf     = (__bf16*)alloc((size_t)B * H * 2);
    float*  c_st     = (float*)alloc((size_t)B * H * 4);
    __bf16* pi_bf    = (__bf16*)alloc((size_t)B * H * 2);
    float*  pmu      = (float*)alloc((size_t)B * Z * 4);
    float*  pvar     = (float*)alloc((size_t)B * Z * 4);
    float*  gates    = (float*)alloc((size_t)B * FH * 4);

    auto cvt = [&](const float* src, __bf16* dst, size_t n) {
        f32_to_bf16_kernel<<<2048, 256, 0, stream>>>(src, dst, n);
    };

    // ---- one-time conversions (re-done every call: deterministic) ----
    cvt(W_q,    wq_bf,    (size_t)H * E);
    cvt(W_qmu,  wqmu_bf,  (size_t)Z * H);
    cvt(W_qvar, wqvar_bf, (size_t)Z * H);
    cvt(W_p,    wp_bf,    (size_t)H * H);
    cvt(W_pmu,  wpmu_bf,  (size_t)Z * H);
    cvt(W_pvar, wpvar_bf, (size_t)Z * H);
    cvt(W_ih,   wih_bf,   (size_t)FH * E);
    cvt(W_hh,   whh_bf,   (size_t)FH * H);
    cvt(temb,   temb_bf,  (size_t)Z * E);
    cvt(rnn_w,  rnnw_bf,  (size_t)T * E);
    bias_sum_kernel<<<FH / 256, 256, 0, stream>>>(b_ih, b_hh, bias_g, FH);

    // ---- h-independent batched phase over all T = S*B tokens ----
    gemm_bf16_kernel<1><<<(T / 128) * (H / 128), 128, 0, stream>>>(
        rnnw_bf, wq_bf, qi_bf, b_q, nullptr, T, H, E);
    gemm_bf16_kernel<0><<<(T / 128) * (Z / 128), 128, 0, stream>>>(
        qi_bf, wqmu_bf, qmu, b_qmu, nullptr, T, Z, H);
    gemm_bf16_kernel<0><<<(T / 128) * (Z / 128), 128, 0, stream>>>(
        qi_bf, wqvar_bf, qvar, b_qvar, nullptr, T, Z, H);
    z_argmax_kernel<<<T, 256, 0, stream>>>(qmu, qvar, eps, z_out, idx);
    gather_kernel<<<T, 128, 0, stream>>>(temb_bf, idx, X_bf);
    gemm_bf16_kernel<0><<<(T / 128) * (FH / 128), 128, 0, stream>>>(
        X_bf, wih_bf, gatesx, bias_g, nullptr, T, FH, E);

    // ---- recurrent phase: h0 = c0 = 0 ----
    zero_u32_kernel<<<256, 256, 0, stream>>>((unsigned*)h_bf, (size_t)B * H * 2 / 4);
    zero_u32_kernel<<<256, 256, 0, stream>>>((unsigned*)c_st, (size_t)B * H);

    for (int t = 0; t < S; t++) {
        gemm_bf16_kernel<1><<<(B / 128) * (H / 128), 128, 0, stream>>>(
            h_bf, wp_bf, pi_bf, b_p, nullptr, B, H, H);
        gemm_bf16_kernel<0><<<(B / 128) * (Z / 128), 128, 0, stream>>>(
            pi_bf, wpmu_bf, pmu, b_pmu, nullptr, B, Z, H);
        gemm_bf16_kernel<0><<<(B / 128) * (Z / 128), 128, 0, stream>>>(
            pi_bf, wpvar_bf, pvar, b_pvar, nullptr, B, Z, H);
        kl_kernel<<<B, 256, 0, stream>>>(
            qmu + (size_t)t * B * Z, qvar + (size_t)t * B * Z, pmu, pvar,
            kl_out + (size_t)t * B);
        gemm_bf16_kernel<2><<<(B / 128) * (FH / 128), 128, 0, stream>>>(
            h_bf, whh_bf, gates, nullptr, gatesx + (size_t)t * B * FH, B, FH, H);
        lstm_kernel<<<(B * H) / 256, 256, 0, stream>>>(gates, c_st, h_bf);
    }
}
